// Attention_20117626814630
// MI455X (gfx1250) — matrix operand order repaired and checked by reading, not compiled, not run on hardware
//
#include <hip/hip_runtime.h>

// ---------------------------------------------------------------------------
// GQA flash-decode for MI455X (gfx1250, wave32).
//
// Roofline: ~1.07 GFLOP vs ~400-536 MB of K/V traffic => ~2 FLOP/byte, i.e.
// HBM-bound (floor ~17-23us @ 23.3 TB/s). Strategy: read every K/V byte once
// with coalesced loads straight into WMMA fragment layouts, keep 2048 waves
// (256 WGs x 8) streaming, mark the single-use K/V stream non-temporal so it
// does not thrash the 192MB L2, software-pipeline loads 2 chunks deep, and
// keep VGPRs < 256 (32-bit cache offsets, no per-token pointer selects) so
// occupancy stays high.
//
// The newest token (scattered to slot_mapping[b] in the reference) is NOT
// written into the cache (inputs must not be mutated). The main loop reads
// the stale cache row; the final reduction applies the exact softmax patch
//   softmax(S u {stale}) - stale-term + fresh-term = softmax(S u {fresh}).
//
// Transposed formulation:
//   S^T[tok,g] = K[tok,d] x Q^T[d,g]      (A = K tile, B = Q^T)
//   O^T[d,g]  += V^T[d,tok] x P^T[tok,g]  (A = V^T,   B = P^T)
// The D-layout of GEMM1 (lane = g column, 8 token rows) maps onto the
// B-layout of GEMM2 with a single shfl_xor(16) to swap token halves.
// ---------------------------------------------------------------------------

#define B_    32
#define H_    32
#define HKV_  8
#define D_    128
#define L_    2048
#define G_    4      // H / HKV
#define NW    8      // waves per workgroup
#define TILE  32     // tokens per tile (two 16x16x32 score WMMAs)
#define ROW_  (HKV_ * D_)   // floats per cache slot

typedef __attribute__((ext_vector_type(16))) _Float16 v16h;
typedef __attribute__((ext_vector_type(8)))  float    v8f;
typedef __attribute__((ext_vector_type(4)))  float    f32x4;

__device__ __forceinline__ v8f wmma16(v16h a, v16h b, v8f c) {
  // D = A x B + C  (f16 inputs, f32 accumulate) -> v_wmma_f32_16x16x32_f16
  return __builtin_amdgcn_wmma_f32_16x16x32_f16(
      /*neg_a=*/false, a, /*neg_b=*/false, b,
      /*c_mod=*/(short)0, c, /*reuse_a=*/false, /*reuse_b=*/false);
}

__device__ __forceinline__ unsigned pk2(float a, float b) {
  union { _Float16 h[2]; unsigned u; } t;
  t.h[0] = (_Float16)a; t.h[1] = (_Float16)b; return t.u;
}

// streaming (single-use) loads: non-temporal so the 400MB K/V pass does not
// evict useful lines from L2
__device__ __forceinline__ f32x4 ldnt4(const float* p) {
  return __builtin_nontemporal_load((const f32x4*)p);
}
__device__ __forceinline__ float ldnt1(const float* p) {
  return __builtin_nontemporal_load(p);
}

__global__ __launch_bounds__(256) void decode_attn_kernel(
    const float* __restrict__ q,          // [B,H,D]
    const float* __restrict__ knew,       // [B,HKV,D]
    const float* __restrict__ vnew,       // [B,HKV,D]
    const float* __restrict__ kc,         // [NUM_SLOTS,HKV,D]
    const float* __restrict__ vc,         // [NUM_SLOTS,HKV,D]
    const long long* __restrict__ slot_mapping,   // [B]
    const long long* __restrict__ active_slots,   // [B,L]
    const long long* __restrict__ context_lens,   // [B]
    float* __restrict__ out)              // [B,H,D]
{
  const int bh   = blockIdx.x;
  const int b    = bh / HKV_;
  const int kv   = bh % HKV_;
  const int tid  = threadIdx.x;
  const int w    = tid >> 5;
  const int lane = tid & 31;
  const int lg   = lane & 15;          // column index (g) / token-in-16
  const bool hi  = lane >= 16;
  const int  h8  = hi ? 8 : 0;

  const int   ctx      = (int)context_lens[b];
  const int   slot_new = (int)slot_mapping[b];
  const float scale    = 0.08838834764831845f;   // 1/sqrt(128)
  const unsigned rowK  = (unsigned)kv * D_;      // kv-head offset within slot

  const float* knew_row = knew + ((size_t)b * HKV_ + kv) * D_;
  const float* vnew_row = vnew + ((size_t)b * HKV_ + kv) * D_;

  __shared__ float ldsO[NW][G_][D_];     // 16 KB
  __shared__ float ldsM[NW][G_];
  __shared__ float ldsL[NW][G_];
  __shared__ float ldsS[2][G_];          // [0]=stale score, [1]=fresh score
  __shared__ int   ldsDoCorr;

  // ---- wave 0: fresh-token softmax-patch scores (s_stale, s_fresh per g).
  // Runs concurrently with other waves' tile work; consumed after the
  // combine barrier. ----
  if (w == 0) {
    int doCorr = 0;
    if (ctx >= 1)
      doCorr = ((int)active_slots[(size_t)b * L_ + (ctx - 1)] == slot_new);
    float ast[G_], afr[G_];
#pragma unroll
    for (int g = 0; g < G_; ++g) { ast[g] = 0.f; afr[g] = 0.f; }
    const float* kst = kc + (size_t)slot_new * ROW_ + rowK;
#pragma unroll
    for (int it = 0; it < D_ / 32; ++it) {
      const int d = it * 32 + lane;
      const float kv_st = kst[d];
      const float kv_fr = knew_row[d];
#pragma unroll
      for (int g = 0; g < G_; ++g) {
        const float qv = q[((size_t)b * H_ + (size_t)kv * G_ + g) * D_ + d];
        ast[g] += qv * kv_st;
        afr[g] += qv * kv_fr;
      }
    }
#pragma unroll
    for (int off = 16; off > 0; off >>= 1)
#pragma unroll
      for (int g = 0; g < G_; ++g) {
        ast[g] += __shfl_xor(ast[g], off, 32);
        afr[g] += __shfl_xor(afr[g], off, 32);
      }
    if (lane == 0) {
#pragma unroll
      for (int g = 0; g < G_; ++g) {
        ldsS[0][g] = ast[g] * scale;
        ldsS[1][g] = afr[g] * scale;
      }
      ldsDoCorr = doCorr;
    }
  }

  // ---- Q^T fragments (B operand), scale folded in. B-layout (16-bit,32x16):
  // elems 0..7 = K(d)=c*32+h8+0..7 ; elems 8..15 = K(d)=c*32+16+h8+0..7.
  const int gq = (lg < G_) ? lg : (G_ - 1);
  const float* qrow = q + ((size_t)b * H_ + (size_t)kv * G_ + gq) * D_;
  v16h qf[4];
#pragma unroll
  for (int c = 0; c < 4; ++c) {
    float tq[16];
    if (lg < G_) {
      const float4* p = (const float4*)(qrow + c * 32 + h8);
      float4 a0 = p[0], a1 = p[1], a2 = p[4], a3 = p[5];
      tq[0]=a0.x; tq[1]=a0.y; tq[2]=a0.z; tq[3]=a0.w;
      tq[4]=a1.x; tq[5]=a1.y; tq[6]=a1.z; tq[7]=a1.w;
      tq[8]=a2.x; tq[9]=a2.y; tq[10]=a2.z; tq[11]=a2.w;
      tq[12]=a3.x; tq[13]=a3.y; tq[14]=a3.z; tq[15]=a3.w;
    } else {
#pragma unroll
      for (int e = 0; e < 16; ++e) tq[e] = 0.f;
    }
#pragma unroll
    for (int e = 0; e < 16; ++e) qf[c][e] = (_Float16)(tq[e] * scale);
  }

  // ---- online-softmax state; O^T accumulators: 8 d-chunks x v8f ----
  float m = -1e30f, lsum = 0.f;
  v8f o[8];
#pragma unroll
  for (int c = 0; c < 8; ++c)
#pragma unroll
    for (int i = 0; i < 8; ++i) o[c][i] = 0.f;

  const int ntiles = (ctx + TILE - 1) / TILE;

  // prefetch the first tile's slot row (one slot per lane)
  int tile = w;
  int slot_pref = (tile < ntiles)
      ? (int)active_slots[(size_t)b * L_ + tile * TILE + lane] : 0;

  for (; tile < ntiles; tile += NW) {
    const int tb = tile * TILE;
    const int slot_lane = slot_pref;
    // prefetch next tile's slots now; the int64 gather latency is hidden
    // behind this tile's entire compute
    const int tnext = tile + NW;
    if (tnext < ntiles)
      slot_pref = (int)active_slots[(size_t)b * L_ + tnext * TILE + lane];

    // ---- K row offsets (32-bit element offsets into kc) ----
    const int s0l = __shfl(slot_lane, lg, 32);
    const int s1l = __shfl(slot_lane, 16 + lg, 32);
    const unsigned kr0o = (unsigned)s0l * ROW_ + rowK;
    const unsigned kr1o = (unsigned)s1l * ROW_ + rowK;

    // ---- V^T row offsets (A operand of GEMM2): lane = one d within chunk,
    // elems 0..15 = tokens {2*h8 .. 2*h8+15}.
    // Computed BEFORE GEMM1 so the first V loads overlap the K stream. ----
    unsigned vro[16];
#pragma unroll
    for (int e = 0; e < 16; ++e) {
      const int ti = 2 * h8 + e;   // token-in-tile
      const int sv = __shfl(slot_lane, ti, 32);
      vro[e] = (unsigned)sv * ROW_ + rowK + (unsigned)lg;
    }
    // preload V chunks 0 and 1 (32 loads in flight across GEMM1 + softmax)
    float vb0[16], vb1[16];
#pragma unroll
    for (int e = 0; e < 16; ++e) vb0[e] = ldnt1(vc + vro[e] + 0 * 16);
#pragma unroll
    for (int e = 0; e < 16; ++e) vb1[e] = ldnt1(vc + vro[e] + 1 * 16);

    // ---- GEMM1: S^T halves. A-layout (16-bit,16x32): lane M=lane&15,
    // elems 0..7 = d{c*32 + h8 + 0..7}, elems 8..15 = d{c*32 + 16 + h8 ..}.
    v8f sc0, sc1;
#pragma unroll
    for (int i = 0; i < 8; ++i) { sc0[i] = 0.f; sc1[i] = 0.f; }
#pragma unroll
    for (int c = 0; c < 4; ++c) {
      const int db = c * 32 + h8;
      {
        f32x4 a0 = ldnt4(kc + kr0o + db), a1 = ldnt4(kc + kr0o + db + 4);
        f32x4 a2 = ldnt4(kc + kr0o + db + 16), a3 = ldnt4(kc + kr0o + db + 20);
        v16h kf;
#pragma unroll
        for (int i = 0; i < 4; ++i) {
          kf[i] = (_Float16)a0[i]; kf[4 + i] = (_Float16)a1[i];
          kf[8 + i] = (_Float16)a2[i]; kf[12 + i] = (_Float16)a3[i];
        }
        sc0 = wmma16(kf, qf[c], sc0);
      }
      {
        f32x4 a0 = ldnt4(kc + kr1o + db), a1 = ldnt4(kc + kr1o + db + 4);
        f32x4 a2 = ldnt4(kc + kr1o + db + 16), a3 = ldnt4(kc + kr1o + db + 20);
        v16h kf;
#pragma unroll
        for (int i = 0; i < 4; ++i) {
          kf[i] = (_Float16)a0[i]; kf[4 + i] = (_Float16)a1[i];
          kf[8 + i] = (_Float16)a2[i]; kf[12 + i] = (_Float16)a3[i];
        }
        sc1 = wmma16(kf, qf[c], sc1);
      }
    }

    // ---- mask + online softmax (per lane: one g column, 16 token scores;
    // partner lane (xor 16) holds the same column's other 16 tokens) ----
    const int rel = ctx - tb - h8;       // element i valid iff i < rel (sc0)
    float p0[8], p1[8];
    float tmax = -1e30f;
#pragma unroll
    for (int i = 0; i < 8; ++i) {
      float a = (i < rel) ? sc0[i] : -1e30f;
      float c2 = (i + 16 < rel) ? sc1[i] : -1e30f;
      p0[i] = a; p1[i] = c2;
      tmax = fmaxf(tmax, fmaxf(a, c2));
    }
    tmax = fmaxf(tmax, __shfl_xor(tmax, 16, 32));
    const float mN    = fmaxf(m, tmax);
    const float alpha = __expf(m - mN);
    float rs = 0.f;
#pragma unroll
    for (int i = 0; i < 8; ++i) {
      p0[i] = __expf(p0[i] - mN);
      p1[i] = __expf(p1[i] - mN);
      rs += p0[i] + p1[i];
    }
    rs += __shfl_xor(rs, 16, 32);
    lsum = lsum * alpha + rs;
    m = mN;
#pragma unroll
    for (int c = 0; c < 8; ++c)
#pragma unroll
      for (int i = 0; i < 8; ++i) o[c][i] *= alpha;

    // ---- P^T fragment (B operand of GEMM2): lane<16 needs tokens 0..15
    // (own p0 + partner p0), lane>=16 needs tokens 16..31 (partner p1 + own)
    unsigned u0[4], u1[4], x0[4], x1[4];
#pragma unroll
    for (int i = 0; i < 4; ++i) {
      u0[i] = pk2(p0[2*i], p0[2*i+1]);
      u1[i] = pk2(p1[2*i], p1[2*i+1]);
    }
#pragma unroll
    for (int i = 0; i < 4; ++i) {
      x0[i] = __shfl_xor((int)u0[i], 16, 32);
      x1[i] = __shfl_xor((int)u1[i], 16, 32);
    }
    union { v16h h; unsigned u[8]; } pf;
#pragma unroll
    for (int i = 0; i < 4; ++i) {
      pf.u[i]     = hi ? x1[i] : u0[i];
      pf.u[4 + i] = hi ? u1[i] : x0[i];
    }

    // ---- GEMM2 over 8 d-chunks, V loads pipelined 2 chunks deep ----
#pragma unroll
    for (int c = 0; c < 8; ++c) {
      v16h vf;
      if ((c & 1) == 0) {
#pragma unroll
        for (int e = 0; e < 16; ++e) vf[e] = (_Float16)vb0[e];
        if (c + 2 < 8)
#pragma unroll
          for (int e = 0; e < 16; ++e) vb0[e] = ldnt1(vc + vro[e] + (c + 2) * 16);
      } else {
#pragma unroll
        for (int e = 0; e < 16; ++e) vf[e] = (_Float16)vb1[e];
        if (c + 2 < 8)
#pragma unroll
          for (int e = 0; e < 16; ++e) vb1[e] = ldnt1(vc + vro[e] + (c + 2) * 16);
      }
      o[c] = wmma16(vf, pf.h, o[c]);     // O^T[d,g] += V^T x P^T
    }
  }

  // ---- combine the 8 waves' partials through LDS ----
  if (lg < G_) {
#pragma unroll
    for (int c = 0; c < 8; ++c)
#pragma unroll
      for (int i = 0; i < 8; ++i)
        ldsO[w][lg][c * 16 + h8 + i] = o[c][i];  // D-layout: lane<16 rows 0..7
    if (!hi) { ldsM[w][lg] = m; ldsL[w][lg] = lsum; }
  }
  __syncthreads();

  const int doC = ldsDoCorr;
  for (int idx = tid; idx < G_ * D_; idx += 256) {
    const int g = idx >> 7;
    const int d = idx & (D_ - 1);
    float gm = -1e30f;
#pragma unroll
    for (int w2 = 0; w2 < NW; ++w2) gm = fmaxf(gm, ldsM[w2][g]);
    float num = 0.f, den = 0.f;
#pragma unroll
    for (int w2 = 0; w2 < NW; ++w2) {
      const float e = __expf(ldsM[w2][g] - gm);
      num += e * ldsO[w2][g][d];
      den += e * ldsL[w2][g];
    }
    if (doC) {
      // exact softmax patch: remove stale-cache token term, add fresh term
      const float sst = ldsS[0][g];
      const float sfr = ldsS[1][g];
      const float M   = fmaxf(gm, sfr);
      const float eg  = __expf(gm - M);
      const float est = __expf(sst - M);
      const float efr = __expf(sfr - M);
      const float vst = vc[(size_t)slot_new * ROW_ + rowK + d];
      const float vfr = vnew_row[d];
      num = num * eg - est * vst + efr * vfr;
      den = den * eg - est + efr;
    }
    out[((size_t)b * H_ + (size_t)kv * G_ + g) * D_ + d] = num / den;
  }
}

extern "C" void kernel_launch(void* const* d_in, const int* in_sizes, int n_in,
                              void* d_out, int out_size, void* d_ws, size_t ws_size,
                              hipStream_t stream) {
  (void)in_sizes; (void)n_in; (void)out_size; (void)d_ws; (void)ws_size;
  const float*     q    = (const float*)d_in[0];
  const float*     knew = (const float*)d_in[1];
  const float*     vnew = (const float*)d_in[2];
  const float*     kc   = (const float*)d_in[3];
  const float*     vc   = (const float*)d_in[4];
  const long long* sm   = (const long long*)d_in[5];
  const long long* as   = (const long long*)d_in[6];
  const long long* cl   = (const long long*)d_in[7];
  float*           out  = (float*)d_out;

  decode_attn_kernel<<<dim3(B_ * HKV_), dim3(NW * 32), 0, stream>>>(
      q, knew, vnew, kc, vc, sm, as, cl, out);
}
